// Co_Pam_Module_56444460204528
// MI455X (gfx1250) — compile-verified, matrix operand order checked by reading, not run on hardware
//
#include <hip/hip_runtime.h>
#include <hip/hip_bf16.h>

typedef __attribute__((ext_vector_type(16))) _Float16 v16h;
typedef __attribute__((ext_vector_type(8)))  float    v8f;

// Fragment K-index mapping for 16-bit WMMA operands (ISA 7.12.2):
// lane<16 : K in {0..7, 16..23};  lane>=16 : K in {8..15, 24..31}
__device__ __forceinline__ int kmap(int ls, int e) {
    return (e & 7) + ((ls >= 16) ? 8 : 0) + ((e >= 8) ? 16 : 0);
}

// ---------------------------------------------------------------------------
// Kernel A1: q = wq@y+bq, k = wk@y+bk, written as swizzled f16 WMMA panels.
// Qswz: A-fragments [b][ib(128)][lane(32)][e(16)]  value = q[d=K(lane,e)][s=ib*16+lane%16]
// Kswz: B-fragments [b][jt(128)][lane(32)][e(16)]  value = k[d=K(lane,e)][t=jt*16+lane%16]
// grid = B*128 blocks, 256 threads
// ---------------------------------------------------------------------------
__global__ __launch_bounds__(256) void pam_qk(const float* __restrict__ y,
                                              const float* __restrict__ wq,
                                              const float* __restrict__ bq,
                                              const float* __restrict__ wk,
                                              const float* __restrict__ bk,
                                              _Float16* __restrict__ Qswz,
                                              _Float16* __restrict__ Kswz) {
    __shared__ float ycol[256 * 16];   // [cy][ss]
    const int b  = blockIdx.x >> 7;
    const int ib = blockIdx.x & 127;
    const int tid = threadIdx.x;
    const int s0 = ib * 16;
    #pragma unroll
    for (int it = 0; it < 16; ++it) {
        int idx = tid + (it << 8);                 // 0..4095 = cy*16+ss
        int cy = idx >> 4, ss = idx & 15;
        ycol[idx] = y[((size_t)(b * 256 + cy)) * 2048 + s0 + ss];
    }
    __syncthreads();
    #pragma unroll
    for (int sl = 0; sl < 4; ++sl) {
        int slot  = tid + (sl << 8);               // 0..1023
        int which = slot >> 9;                     // 0=q, 1=k
        int rem   = slot & 511;
        int ls = rem >> 4, e = rem & 15;
        int dK = kmap(ls, e);
        int sc = ls & 15;
        const float* w  = which ? wk : wq;
        const float* bb = which ? bk : bq;
        float acc = bb[dK];
        for (int cy = 0; cy < 256; ++cy)
            acc += w[dK * 256 + cy] * ycol[cy * 16 + sc];
        _Float16* dst = which ? Kswz : Qswz;
        dst[((size_t)(b * 128 + ib)) * 512 + rem] = (_Float16)acc;
    }
}

// ---------------------------------------------------------------------------
// Kernel A2: v_pool = wv @ pool4(x) + 4*bv, written as swizzled f16 A-panels.
// Vswz: [b][mt(4)][kt(64)][lane(32)][e(16)] value = v_pool[m=mt*16+lane%16][j=kt*32+K(lane,e)]
// grid = B*64 blocks, 256 threads
// ---------------------------------------------------------------------------
__global__ __launch_bounds__(256) void pam_v(const float* __restrict__ x,
                                             const float* __restrict__ wv,
                                             const float* __restrict__ bv,
                                             _Float16* __restrict__ Vswz) {
    __shared__ float xp[64 * 32];     // pooled x: [c'][jj]
    __shared__ float wvs[64 * 64];
    const int b  = blockIdx.x >> 6;
    const int kt = blockIdx.x & 63;
    const int tid = threadIdx.x;
    #pragma unroll
    for (int it = 0; it < 8; ++it) {
        int slot = tid + (it << 8);                // 0..2047 = c*32+jj
        int c = slot >> 5, jj = slot & 31;
        size_t base = ((size_t)(b * 64 + c)) * 8192 + (size_t)kt * 128 + jj * 4;
        xp[slot] = x[base] + x[base + 1] + x[base + 2] + x[base + 3];
    }
    #pragma unroll
    for (int it = 0; it < 16; ++it) {
        int idx = tid + (it << 8);
        wvs[idx] = wv[idx];
    }
    __syncthreads();
    #pragma unroll
    for (int it = 0; it < 8; ++it) {
        int slot = tid + (it << 8);                // 0..2047
        int mt = slot >> 9, rem = slot & 511;
        int ls = rem >> 4, e = rem & 15;
        int m = mt * 16 + (ls & 15);
        int k = kmap(ls, e);
        float acc = 4.0f * bv[m];
        for (int c = 0; c < 64; ++c)
            acc += wvs[m * 64 + c] * xp[c * 32 + k];
        Vswz[(((size_t)(b * 4 + mt)) * 64 + kt) * 512 + rem] = (_Float16)acc;
    }
}

// ---------------------------------------------------------------------------
// Kernel B: fused attention. One WG = (batch b, 16-row i-block ib).
// LDS layout (dynamic): [0,128K) E f32 16x2048 (reused as Opart),
//                       [128K,192K) P f16 swizzled B-fragments, then rowsum[16].
// ---------------------------------------------------------------------------
#define LDS_E_BYTES   (16 * 2048 * 4)
#define LDS_P_BYTES   (16 * 2048 * 2)
#define LDS_TOTAL     (LDS_E_BYTES + LDS_P_BYTES + 64)

__global__ __launch_bounds__(256) void pam_attn(const _Float16* __restrict__ Qswz,
                                                const _Float16* __restrict__ Kswz,
                                                const _Float16* __restrict__ Vswz,
                                                float* __restrict__ Obuf) {
    extern __shared__ char smem[];
    float*    E      = (float*)smem;                          // 16 x 2048
    _Float16* P      = (_Float16*)(smem + LDS_E_BYTES);       // swizzled exp(E-max)
    float*    rowsum = (float*)(smem + LDS_E_BYTES + LDS_P_BYTES);

    const int b    = blockIdx.x >> 7;
    const int ib   = blockIdx.x & 127;
    const int tid  = threadIdx.x;
    const int wave = tid >> 5;
    const int lane = tid & 31;

    // ---- Stage 1: energy E[i(16), j(2048)] = q^T k  (K=32, one WMMA per 16x16 tile)
    v16h qa = *(const v16h*)(Qswz + ((size_t)(b * 128 + ib)) * 512 + lane * 16);
    for (int t = 0; t < 16; ++t) {
        int jt = wave * 16 + t;                                // 0..127
        v16h kb = *(const v16h*)(Kswz + ((size_t)(b * 128 + jt)) * 512 + lane * 16);
        v8f c = {};
        c = __builtin_amdgcn_wmma_f32_16x16x32_f16(false, qa, false, kb,
                                                   (short)0, c, false, false);
        int n = jt * 16 + (lane & 15);
        int mbase = (lane < 16) ? 0 : 8;
        #pragma unroll
        for (int r = 0; r < 8; ++r)
            E[(mbase + r) * 2048 + n] = c[r];
    }
    __syncthreads();

    // ---- Stage 2: softmax rows (unnormalized exp stored into swizzled f16 B-frags)
    #pragma unroll
    for (int rr = 0; rr < 2; ++rr) {
        int row = wave * 2 + rr;                               // 0..15
        float m = -1e30f;
        for (int t = 0; t < 64; ++t)
            m = fmaxf(m, E[row * 2048 + lane + (t << 5)]);
        for (int off = 16; off; off >>= 1)
            m = fmaxf(m, __shfl_xor(m, off, 32));
        float s = 0.0f;
        for (int t = 0; t < 64; ++t) {
            int j = lane + (t << 5);
            float p = __expf(E[row * 2048 + j] - m);
            s += p;
            int kt = j >> 5, kl = j & 31;
            int sub = (kl >> 3) & 1;
            int e   = (kl & 7) + ((kl >> 4) << 3);
            P[kt * 512 + (row + (sub << 4)) * 16 + e] = (_Float16)p;
        }
        for (int off = 16; off; off >>= 1)
            s += __shfl_xor(s, off, 32);
        if (lane == 0) rowsum[row] = s;
    }
    __syncthreads();

    // ---- Stage 3: O[c(64), i(16)] = v_pool @ P^T ; each wave owns a K-slice of 256 j
    v8f a0 = {}, a1 = {}, a2 = {}, a3 = {};
    for (int t = 0; t < 8; ++t) {
        int kt = wave * 8 + t;                                 // 0..63
        v16h pb = *(const v16h*)(P + kt * 512 + lane * 16);
        v16h va0 = *(const v16h*)(Vswz + (((size_t)(b * 4 + 0)) * 64 + kt) * 512 + lane * 16);
        v16h va1 = *(const v16h*)(Vswz + (((size_t)(b * 4 + 1)) * 64 + kt) * 512 + lane * 16);
        v16h va2 = *(const v16h*)(Vswz + (((size_t)(b * 4 + 2)) * 64 + kt) * 512 + lane * 16);
        v16h va3 = *(const v16h*)(Vswz + (((size_t)(b * 4 + 3)) * 64 + kt) * 512 + lane * 16);
        a0 = __builtin_amdgcn_wmma_f32_16x16x32_f16(false, va0, false, pb, (short)0, a0, false, false);
        a1 = __builtin_amdgcn_wmma_f32_16x16x32_f16(false, va1, false, pb, (short)0, a1, false, false);
        a2 = __builtin_amdgcn_wmma_f32_16x16x32_f16(false, va2, false, pb, (short)0, a2, false, false);
        a3 = __builtin_amdgcn_wmma_f32_16x16x32_f16(false, va3, false, pb, (short)0, a3, false, false);
    }
    // store per-wave partials (reuse E region), then cross-wave reduce
    float* Opart = E;   // [wave][mt][lane][r] : 8*4*32*8 f32 = 32 KB
    #pragma unroll
    for (int r = 0; r < 8; ++r) {
        Opart[(((wave * 4 + 0) * 32 + lane) << 3) + r] = a0[r];
        Opart[(((wave * 4 + 1) * 32 + lane) << 3) + r] = a1[r];
        Opart[(((wave * 4 + 2) * 32 + lane) << 3) + r] = a2[r];
        Opart[(((wave * 4 + 3) * 32 + lane) << 3) + r] = a3[r];
    }
    __syncthreads();
    for (int idx = tid; idx < 1024; idx += 256) {
        int c = idx >> 4, i = idx & 15;
        int mt = c >> 4, cl = c & 15;
        int r = cl & 7, sub = cl >> 3;
        int l2 = i + (sub << 4);
        float s = 0.0f;
        #pragma unroll
        for (int w = 0; w < 8; ++w)
            s += Opart[(((w * 4 + mt) * 32 + l2) << 3) + r];
        Obuf[((size_t)(b * 64 + c)) * 2048 + ib * 16 + i] = s / rowsum[i];
    }
}

// ---------------------------------------------------------------------------
// Kernel C: per-channel BN statistics over g*O (repeat-x4 preserves mean/var).
// grid = 64 blocks (one per channel), 256 threads.  stats[c] = {scale*g, shift}
// ---------------------------------------------------------------------------
__global__ __launch_bounds__(256) void pam_stats(const float* __restrict__ Obuf,
                                                 const float* __restrict__ gamma,
                                                 const float* __restrict__ bn_w,
                                                 const float* __restrict__ bn_b,
                                                 float* __restrict__ stats) {
    __shared__ float r1[256], r2[256];
    const int c = blockIdx.x, tid = threadIdx.x;
    const float g = gamma[0];
    float s1 = 0.0f, s2 = 0.0f;
    for (int b = 0; b < 8; ++b)
        for (int i = tid; i < 2048; i += 256) {
            float v = g * Obuf[((size_t)(b * 64 + c)) * 2048 + i];
            s1 += v; s2 += v * v;
        }
    r1[tid] = s1; r2[tid] = s2; __syncthreads();
    for (int off = 128; off; off >>= 1) {
        if (tid < off) { r1[tid] += r1[tid + off]; r2[tid] += r2[tid + off]; }
        __syncthreads();
    }
    if (tid == 0) {
        float mean = r1[0] * (1.0f / 16384.0f);
        float var  = r2[0] * (1.0f / 16384.0f) - mean * mean;
        float rs   = rsqrtf(var + 1e-5f);
        float sc   = bn_w[c] * rs;
        stats[c * 2 + 0] = sc * g;                 // applied to raw O
        stats[c * 2 + 1] = bn_b[c] - mean * sc;    // mean already includes g
    }
}

// ---------------------------------------------------------------------------
// Kernel D: out = x + scale[c]*O[b,c,sx/4] + shift[c]   (4M elems, BW-bound)
// ---------------------------------------------------------------------------
__global__ __launch_bounds__(256) void pam_final(const float* __restrict__ x,
                                                 const float* __restrict__ Obuf,
                                                 const float* __restrict__ stats,
                                                 float* __restrict__ out) {
    size_t idx = (size_t)blockIdx.x * 256 + threadIdx.x;   // < 4194304
    int sx = (int)(idx & 8191);
    int c  = (int)((idx >> 13) & 63);
    int b  = (int)(idx >> 19);
    float o = Obuf[((size_t)(b * 64 + c)) * 2048 + (sx >> 2)];
    out[idx] = x[idx] + stats[c * 2 + 0] * o + stats[c * 2 + 1];
}

// ---------------------------------------------------------------------------
extern "C" void kernel_launch(void* const* d_in, const int* in_sizes, int n_in,
                              void* d_out, int out_size, void* d_ws, size_t ws_size,
                              hipStream_t stream) {
    (void)in_sizes; (void)n_in; (void)out_size; (void)ws_size;
    const float* x     = (const float*)d_in[0];
    const float* y     = (const float*)d_in[1];
    const float* wq    = (const float*)d_in[2];
    const float* bq    = (const float*)d_in[3];
    const float* wk    = (const float*)d_in[4];
    const float* bk    = (const float*)d_in[5];
    const float* wv    = (const float*)d_in[6];
    const float* bv    = (const float*)d_in[7];
    const float* gamma = (const float*)d_in[8];
    const float* bn_w  = (const float*)d_in[9];
    const float* bn_b  = (const float*)d_in[10];
    float* out = (float*)d_out;
    char*  ws  = (char*)d_ws;

    _Float16* Qswz = (_Float16*)(ws);                 // 1 MB : 8*128*512 f16
    _Float16* Kswz = (_Float16*)(ws + (1u << 20));    // 1 MB
    _Float16* Vswz = (_Float16*)(ws + (2u << 20));    // 2 MB : 8*4*64*512 f16
    float*    Obuf = (float*)(ws + (4u << 20));       // 4 MB : 8*64*2048 f32
    float*    stats= (float*)(ws + (8u << 20));       // 512 B

    pam_qk<<<1024, 256, 0, stream>>>(y, wq, bq, wk, bk, Qswz, Kswz);
    pam_v<<<512, 256, 0, stream>>>(x, wv, bv, Vswz);

    hipFuncSetAttribute(reinterpret_cast<const void*>(&pam_attn),
                        hipFuncAttributeMaxDynamicSharedMemorySize, LDS_TOTAL);
    pam_attn<<<1024, 256, LDS_TOTAL, stream>>>(Qswz, Kswz, Vswz, Obuf);

    pam_stats<<<64, 256, 0, stream>>>(Obuf, gamma, bn_w, bn_b, stats);
    pam_final<<<16384, 256, 0, stream>>>(x, Obuf, stats, out);
}
